// ParallelTransformerBlock_27642409517467
// MI455X (gfx1250) — matrix operand order repaired and checked by reading, not compiled, not run on hardware
//
#include <hip/hip_runtime.h>
#include <hip/hip_bf16.h>

// ---------------------------------------------------------------------------
// Model dims (fixed by the reference)
// ---------------------------------------------------------------------------
#define D_MODEL   2048
#define NUM_HEADS 16
#define HEAD      128
#define FAN       4
#define QKV_OUT   (3 * D_MODEL + FAN * D_MODEL)   // 14336
#define HID_DIM   (FAN * D_MODEL)                 // 8192
#define BATCH     2
#define SEQ       2048
#define ROWS      (BATCH * SEQ)                   // 4096

typedef __attribute__((ext_vector_type(16))) __bf16 v16bf;
typedef __attribute__((ext_vector_type(8)))  float  v8f;

union Frag16 {
    v16bf h;
    uint4 q[2];
    unsigned int u[8];
};

static __device__ __forceinline__ unsigned short f32_to_bf16(float f) {
    unsigned int u = __float_as_uint(f);
    u += 0x7FFFu + ((u >> 16) & 1u);   // round-to-nearest-even
    return (unsigned short)(u >> 16);
}

static __device__ __forceinline__ unsigned pack_bf16x2(float a, float b) {
    return (unsigned)f32_to_bf16(a) | ((unsigned)f32_to_bf16(b) << 16);
}

static __device__ __forceinline__ v8f wmma_bf16(v16bf a, v16bf b, v8f c) {
    // D = A(16x32 bf16) * B(32x16 bf16) + C(16x16 f32)
    return __builtin_amdgcn_wmma_f32_16x16x32_bf16(
        false, a, false, b, (short)0, c, false, false);
}

// Generic LDS pointer -> 32-bit LDS byte offset (ISA 10.2: LDS aperture
// addresses truncate to the low 32 bits).
static __device__ __forceinline__ unsigned lds_off(const void* p) {
    return (unsigned)(size_t)p;
}

static __device__ __forceinline__ void async_b128(unsigned lds_byte,
                                                  const void* gptr) {
    asm volatile("global_load_async_to_lds_b128 %0, %1, off"
                 :: "v"(lds_byte), "v"((unsigned long long)gptr) : "memory");
}

// ---------------------------------------------------------------------------
// Kernel 1: LayerNorm over D_MODEL, fp32 -> bf16.  One block per row.
// ---------------------------------------------------------------------------
__global__ __launch_bounds__(256)
void ln_cast_x(const float* __restrict__ x, unsigned short* __restrict__ xln) {
    __shared__ float red[256];
    const int row = blockIdx.x;
    const int tid = threadIdx.x;
    const float* xr = x + (size_t)row * D_MODEL + tid * 8;

    float vals[8];
    float s = 0.f;
#pragma unroll
    for (int i = 0; i < 8; ++i) {
        vals[i] = xr[i];
        s += vals[i];
    }
    red[tid] = s; __syncthreads();
    for (int o = 128; o > 0; o >>= 1) {
        if (tid < o) red[tid] += red[tid + o];
        __syncthreads();
    }
    const float mu = red[0] * (1.f / D_MODEL);
    __syncthreads();

    float vs = 0.f;
#pragma unroll
    for (int i = 0; i < 8; ++i) {
        float d = vals[i] - mu;
        vs += d * d;
    }
    red[tid] = vs; __syncthreads();
    for (int o = 128; o > 0; o >>= 1) {
        if (tid < o) red[tid] += red[tid + o];
        __syncthreads();
    }
    const float inv = rsqrtf(red[0] * (1.f / D_MODEL) + 1e-5f);

    uint4 w;
    w.x = pack_bf16x2((vals[0] - mu) * inv, (vals[1] - mu) * inv);
    w.y = pack_bf16x2((vals[2] - mu) * inv, (vals[3] - mu) * inv);
    w.z = pack_bf16x2((vals[4] - mu) * inv, (vals[5] - mu) * inv);
    w.w = pack_bf16x2((vals[6] - mu) * inv, (vals[7] - mu) * inv);
    *(uint4*)(xln + (size_t)row * D_MODEL + tid * 8) = w;
}

// ---------------------------------------------------------------------------
// Kernel 2: fused transpose + fp32->bf16 cast for weights.
//   src: K x N (fp32, row-major)  ->  dst: N x K (bf16, row-major)
// ---------------------------------------------------------------------------
__global__ __launch_bounds__(256)
void cast_transpose(const float* __restrict__ src, unsigned short* __restrict__ dst,
                    int K, int N) {
    __shared__ float tile[32][33];
    const int bn = blockIdx.x * 32;   // N coordinate
    const int bk = blockIdx.y * 32;   // K coordinate
    const int tx = threadIdx.x & 31;
    const int ty = threadIdx.x >> 5;  // 8 rows per pass
#pragma unroll
    for (int r = 0; r < 32; r += 8)
        tile[ty + r][tx] = src[(size_t)(bk + ty + r) * N + bn + tx];
    __syncthreads();
#pragma unroll
    for (int r = 0; r < 32; r += 8)
        dst[(size_t)(bn + ty + r) * K + bk + tx] = f32_to_bf16(tile[tx][ty + r]);
}

// ---------------------------------------------------------------------------
// Kernel 3: tiled bf16 GEMM with WMMA + async-to-LDS double buffering.
//   C(MxN) = A(MxK) * Bt(NxK)^T + bias[n]  (+ resid)
//   Block tile 128x256, 8 waves (2x4), each wave 64x64 (4x4 accumulators):
//   16 WMMA per 16 ds_load_b128 per K-tile per wave.
// ---------------------------------------------------------------------------
#define BM 128
#define BN 256
#define KT 32
#define SA 40   // LDS row stride in ushorts: 80B = 5*16B -> b128-aligned, conflict-free

// Copy a (CH*64)-row x 32-col bf16 tile into LDS via async DMA.
template <int CH>
static __device__ __forceinline__ void async_copy_tile(
        const unsigned short* __restrict__ g, int gstride,
        unsigned lds_base, int tid) {
#pragma unroll
    for (int it = 0; it < CH; ++it) {
        const int c = tid + it * 256;
        const int row = c >> 2;
        const int col = (c & 3) * 8;
        async_b128(lds_base + (unsigned)((row * SA + col) * 2),
                   g + (size_t)row * gstride + col);
    }
}

static __device__ __forceinline__ v16bf load_frag_a(const unsigned short* lds,
                                                    int rb, int lane) {
    // A-matrix 16x32 bf16, ISA layout: lanes 0-15 M=lane, halves split K.
    const int lr = lane & 15, hb = lane >> 4;
    const unsigned short* rp = lds + (rb + lr) * SA + hb * 8;
    Frag16 f;
    f.q[0] = *(const uint4*)(rp);        // K = hb*8 + 0..7
    f.q[1] = *(const uint4*)(rp + 16);   // K = 16 + hb*8 + 0..7
    return f.h;
}

static __device__ __forceinline__ v16bf load_frag_b(const unsigned short* lds,
                                                    int nb, int lane) {
    // B-matrix 32x16 bf16 held as [n][k]: lane lr -> N, K = hb*8+0..7 and 16+hb*8+0..7.
    const int lr = lane & 15, hb = lane >> 4;
    const unsigned short* rp = lds + (nb + lr) * SA + hb * 8;
    Frag16 f;
    f.q[0] = *(const uint4*)(rp);
    f.q[1] = *(const uint4*)(rp + 16);
    return f.h;
}

template <bool RESID>
__global__ __launch_bounds__(256)
void gemm_bf16(const unsigned short* __restrict__ A,   // M x K (bf16)
               const unsigned short* __restrict__ Bt,  // N x K (bf16, transposed)
               const float* __restrict__ bias,         // [N]
               const float* __restrict__ resid,        // [M x N] (if RESID)
               float* __restrict__ C,                  // [M x N]
               int N, int K) {
    __shared__ unsigned short lsA[2][BM * SA];   // 2 x 10240 B
    __shared__ unsigned short lsB[2][BN * SA];   // 2 x 20480 B

    const int tid  = threadIdx.x;
    const int wid  = tid >> 5;
    const int lane = tid & 31;
    const int bm = blockIdx.y * BM;
    const int bn = blockIdx.x * BN;
    const int wm = (wid >> 2) * 64;   // 2 wave-rows
    const int wn = (wid & 3) * 64;    // 4 wave-cols

    const unsigned aBase0 = lds_off(&lsA[0][0]);
    const unsigned aBase1 = lds_off(&lsA[1][0]);
    const unsigned bBase0 = lds_off(&lsB[0][0]);
    const unsigned bBase1 = lds_off(&lsB[1][0]);

    v8f acc[4][4];
    v8f z = {};
#pragma unroll
    for (int mi = 0; mi < 4; ++mi)
#pragma unroll
        for (int ni = 0; ni < 4; ++ni) acc[mi][ni] = z;

    auto compute = [&](int bufi) {
        const unsigned short* la = lsA[bufi];
        const unsigned short* lb = lsB[bufi];
        v16bf af[4];
#pragma unroll
        for (int mi = 0; mi < 4; ++mi)
            af[mi] = load_frag_a(la, wm + 16 * mi, lane);
#pragma unroll
        for (int ni = 0; ni < 4; ++ni) {
            v16bf bfm = load_frag_b(lb, wn + 16 * ni, lane);
#pragma unroll
            for (int mi = 0; mi < 4; ++mi)
                acc[mi][ni] = wmma_bf16(af[mi], bfm, acc[mi][ni]);
        }
    };

    // Prime buffer 0 with the first K tile.
    async_copy_tile<2>(A  + (size_t)bm * K, K, aBase0, tid);
    async_copy_tile<4>(Bt + (size_t)bn * K, K, bBase0, tid);

    int buf = 0;
    for (int k0 = 0; k0 + KT < K; k0 += KT, buf ^= 1) {
        // Kick the next tile into the other buffer, wait for the current one
        // only (async global loads complete in order: 6 newly issued pending).
        async_copy_tile<2>(A  + (size_t)bm * K + k0 + KT, K,
                           buf ? aBase0 : aBase1, tid);
        async_copy_tile<4>(Bt + (size_t)bn * K + k0 + KT, K,
                           buf ? bBase0 : bBase1, tid);
        asm volatile("s_wait_asynccnt 0x6" ::: "memory");
        __syncthreads();
        compute(buf);
        __syncthreads();   // all waves done before this buffer is re-filled
    }
    asm volatile("s_wait_asynccnt 0x0" ::: "memory");
    __syncthreads();
    compute(buf);

    // ---- epilogue: C layout lanes 0-15 -> M=j, lanes 16-31 -> M=8+j ----
    const int lr = lane & 15, hb = lane >> 4;
#pragma unroll
    for (int mi = 0; mi < 4; ++mi)
#pragma unroll
        for (int ni = 0; ni < 4; ++ni)
#pragma unroll
            for (int j = 0; j < 8; ++j) {
                int row = bm + wm + 16 * mi + j + (hb ? 8 : 0);
                int col = bn + wn + 16 * ni + lr;
                float v = acc[mi][ni][j] + bias[col];
                size_t idx = (size_t)row * N + col;
                if (RESID) v += resid[idx];
                C[idx] = v;
            }
}

// ---------------------------------------------------------------------------
// Kernel 4: per-head LayerNorm (or plain pack for V), fp32 -> bf16 [B,H,T,D]
// ---------------------------------------------------------------------------
__global__ __launch_bounds__(128)
void head_pack(const float* __restrict__ qkvh, unsigned short* __restrict__ dst,
               int col_off, int do_ln) {
    const int wid  = threadIdx.x >> 5;
    const int lane = threadIdx.x & 31;
    const int r = blockIdx.x * 4 + wid;            // B*T*H rows
    const int b = r / (SEQ * NUM_HEADS);
    const int rem = r % (SEQ * NUM_HEADS);
    const int t = rem / NUM_HEADS;
    const int h = rem % NUM_HEADS;

    const float* src = qkvh + (size_t)(b * SEQ + t) * QKV_OUT + col_off + h * HEAD;
    float v[4];
#pragma unroll
    for (int i = 0; i < 4; ++i) v[i] = src[lane * 4 + i];

    if (do_ln) {
        float s = v[0] + v[1] + v[2] + v[3];
#pragma unroll
        for (int m = 16; m >= 1; m >>= 1) s += __shfl_xor(s, m, 32);
        const float mu = s * (1.f / HEAD);
        float vs = 0.f;
#pragma unroll
        for (int i = 0; i < 4; ++i) { float d = v[i] - mu; vs += d * d; }
#pragma unroll
        for (int m = 16; m >= 1; m >>= 1) vs += __shfl_xor(vs, m, 32);
        const float inv = rsqrtf(vs * (1.f / HEAD) + 1e-5f);
#pragma unroll
        for (int i = 0; i < 4; ++i) v[i] = (v[i] - mu) * inv;
    }

    unsigned short* d =
        dst + ((size_t)(b * NUM_HEADS + h) * SEQ + t) * HEAD + lane * 4;
    uint2 w;
    w.x = pack_bf16x2(v[0], v[1]);
    w.y = pack_bf16x2(v[2], v[3]);
    *(uint2*)d = w;
}

// ---------------------------------------------------------------------------
// Kernel 5: exact GELU on hidden, fp32 -> bf16 (paired b32 stores)
// ---------------------------------------------------------------------------
__global__ __launch_bounds__(256)
void gelu_cast(const float* __restrict__ qkvh, unsigned short* __restrict__ hid) {
    long long i = (long long)blockIdx.x * blockDim.x + threadIdx.x;
    long long stride = (long long)gridDim.x * blockDim.x;
    const long long n2 = (long long)ROWS * HID_DIM / 2;
    for (; i < n2; i += stride) {
        long long row = i >> 12;                 // / (HID_DIM/2)
        long long c0  = (i & (HID_DIM / 2 - 1)) * 2;
        const float* s = qkvh + row * QKV_OUT + 3 * D_MODEL + c0;
        float x0 = s[0], x1 = s[1];
        float g0 = 0.5f * x0 * (1.f + erff(x0 * 0.70710678118654752f));
        float g1 = 0.5f * x1 * (1.f + erff(x1 * 0.70710678118654752f));
        *(unsigned*)(hid + row * HID_DIM + c0) = pack_bf16x2(g0, g1);
    }
}

// ---------------------------------------------------------------------------
// Kernel 6: causal flash attention (FA2-style).
//   Block = (b, h, 128 adjacent queries), 8 waves of one 16-query tile each.
//   K chunk staged once per block via async-to-LDS; V staged transposed; row
//   sums come free via an extra P@ones WMMA (no cross-lane reduction).
// ---------------------------------------------------------------------------
#define SKV 136   // K chunk row stride (elems): 272B = 17*16B, conflict-free
#define SVT 40    // Vt row stride
#define SP  40    // per-wave P-tile stride

__global__ __launch_bounds__(256)
void attention(const unsigned short* __restrict__ qh,
               const unsigned short* __restrict__ kh,
               const unsigned short* __restrict__ vh,
               float* __restrict__ out) {
    __shared__ unsigned short lsk[32 * SKV];        // 8704 B
    __shared__ unsigned short lsvt[HEAD * SVT];     // 10240 B
    __shared__ unsigned short pbuf[8 * 16 * SP];    // 10240 B

    const int tid  = threadIdx.x;
    const int wid  = tid >> 5;
    const int lane = tid & 31;
    const int lr = lane & 15, hb = lane >> 4;

    const int bid = blockIdx.x;                 // B*H*(T/128) = 512
    const int b   = bid >> 8;
    const int rem = bid & 255;
    const int h   = rem >> 4;
    const int q0b = (rem & 15) * 128;           // block query base
    const int q0  = q0b + wid * 16;             // this wave's query tile
    const int q0s = __builtin_amdgcn_readfirstlane(q0);  // scalar branch below

    const size_t head_base = (size_t)(b * NUM_HEADS + h) * SEQ * HEAD;
    const unsigned short* qp = qh + head_base;
    const unsigned short* kp = kh + head_base;
    const unsigned short* vp = vh + head_base;
    unsigned short* pw = pbuf + wid * (16 * SP);
    const unsigned kBase = lds_off(&lsk[0]);

    // Q fragments: 4 x (16x32 bf16) straight from global, b128 per lane.
    Frag16 qf[4];
#pragma unroll
    for (int f = 0; f < 4; ++f) {
        const unsigned short* rp = qp + (size_t)(q0 + lr) * HEAD + 32 * f + hb * 8;
        qf[f].q[0] = *(const uint4*)(rp);
        qf[f].q[1] = *(const uint4*)(rp + 16);
    }

    // All-ones B fragment (bf16 1.0 packed) for rowsum-via-WMMA.
    Frag16 ones;
#pragma unroll
    for (int i = 0; i < 8; ++i) ones.u[i] = 0x3F803F80u;

    v8f o[8];
    v8f z = {};
#pragma unroll
    for (int f = 0; f < 8; ++f) o[f] = z;
    float mrow[8], lrow[8];
#pragma unroll
    for (int j = 0; j < 8; ++j) { mrow[j] = -1e30f; lrow[j] = 1.f; }

    const float scale = 0.08838834764831845f;   // 1/sqrt(128)
    const int kend = q0b + 128;                 // causal bound for whole block

    for (int kc = 0; kc < kend; kc += 32) {
        __syncthreads();   // previous chunk fully consumed by all waves

        // ---- stage K chunk 32x128 via async DMA (2 instr / lane) ----
#pragma unroll
        for (int it = 0; it < 2; ++it) {
            const int c = tid + it * 256;
            const int row = c >> 4;
            const int col = (c & 15) * 8;
            async_b128(kBase + (unsigned)((row * SKV + col) * 2),
                       kp + (size_t)(kc + row) * HEAD + col);
        }
        // ---- stage V chunk transposed to [hd][key] (b128 load + 8 b16 ds) ----
#pragma unroll
        for (int it = 0; it < 2; ++it) {
            const int c = tid + it * 256;
            const int row = c >> 4;
            const int col = (c & 15) * 8;
            const uint4 dv = *(const uint4*)(vp + (size_t)(kc + row) * HEAD + col);
            alignas(16) unsigned short tmp[8];
            *(uint4*)tmp = dv;
#pragma unroll
            for (int j = 0; j < 8; ++j) lsvt[(col + j) * SVT + row] = tmp[j];
        }
        asm volatile("s_wait_asynccnt 0x0" ::: "memory");
        __syncthreads();

        if (kc < q0s + 16) {     // scalar condition -> EXEC untouched
            // ---- S = Q Kt (two 16-key blocks) from LDS ----
            v8f s0 = z, s1 = z;
#pragma unroll
            for (int f = 0; f < 4; ++f) {
                Frag16 bk;
                const unsigned short* rp0 = lsk + lr * SKV + 32 * f + hb * 8;
                bk.q[0] = *(const uint4*)(rp0);
                bk.q[1] = *(const uint4*)(rp0 + 16);
                s0 = wmma_bf16(qf[f].h, bk.h, s0);
                const unsigned short* rp1 = rp0 + 16 * SKV;
                bk.q[0] = *(const uint4*)(rp1);
                bk.q[1] = *(const uint4*)(rp1 + 16);
                s1 = wmma_bf16(qf[f].h, bk.h, s1);
            }

            // ---- mask + online max (shuffles only for the max) ----
            float alpha[8];
#pragma unroll
            for (int j = 0; j < 8; ++j) {
                const int qrow = q0 + j + (hb ? 8 : 0);
                float a = (kc + lr      <= qrow) ? s0[j] * scale : -1e30f;
                float c = (kc + 16 + lr <= qrow) ? s1[j] * scale : -1e30f;
                float mx = fmaxf(a, c);
#pragma unroll
                for (int m = 1; m < 16; m <<= 1)
                    mx = fmaxf(mx, __shfl_xor(mx, m, 32));
                const float mnew = fmaxf(mrow[j], mx);
                alpha[j] = expf(mrow[j] - mnew);
                mrow[j] = mnew;
                s0[j] = expf(a - mnew);
                s1[j] = expf(c - mnew);
            }
#pragma unroll
            for (int f = 0; f < 8; ++f)
#pragma unroll
                for (int j = 0; j < 8; ++j) o[f][j] *= alpha[j];

            // ---- C layout -> A layout for P through per-wave LDS tile ----
#pragma unroll
            for (int j = 0; j < 8; ++j) {
                const int rowl = j + (hb ? 8 : 0);
                pw[rowl * SP + lr]      = f32_to_bf16(s0[j]);
                pw[rowl * SP + 16 + lr] = f32_to_bf16(s1[j]);
            }
            asm volatile("s_wait_dscnt 0x0" ::: "memory");   // wave-local fence
            __builtin_amdgcn_wave_barrier();

            Frag16 pa;
            {
                const unsigned short* rp = pw + lr * SP + hb * 8;
                pa.q[0] = *(const uint4*)(rp);
                pa.q[1] = *(const uint4*)(rp + 16);
            }
            asm volatile("" ::: "memory");

            // ---- row sums via one WMMA: lsum[j] = sum_k P[row_j][k] ----
            v8f lsum = wmma_bf16(pa.h, ones.h, z);
#pragma unroll
            for (int j = 0; j < 8; ++j)
                lrow[j] = lrow[j] * alpha[j] + lsum[j];

            // ---- O += P V  (V fragments: 2 x ds_load_b128 from Vt) ----
#pragma unroll
            for (int f = 0; f < 8; ++f) {
                Frag16 vb;
                const unsigned short* rp = lsvt + (16 * f + lr) * SVT + hb * 8;
                vb.q[0] = *(const uint4*)(rp);
                vb.q[1] = *(const uint4*)(rp + 16);
                o[f] = wmma_bf16(pa.h, vb.h, o[f]);
            }
        }
    }

    // lrow started at 1 with mrow=-1e30: the first real chunk multiplies it by
    // alpha=exp(-1e30-m)=0, so the +1 never survives; rows always see key 0.
#pragma unroll
    for (int f = 0; f < 8; ++f)
#pragma unroll
        for (int j = 0; j < 8; ++j) {
            const int qrow = q0 + j + (hb ? 8 : 0);
            const size_t idx =
                ((size_t)(b * SEQ + qrow)) * D_MODEL + h * HEAD + 16 * f + lr;
            out[idx] += o[f][j] / lrow[j];
        }
}

// ---------------------------------------------------------------------------
// Host-side launcher
// ---------------------------------------------------------------------------
extern "C" void kernel_launch(void* const* d_in, const int* in_sizes, int n_in,
                              void* d_out, int out_size, void* d_ws, size_t ws_size,
                              hipStream_t stream) {
    (void)in_sizes; (void)n_in; (void)out_size; (void)ws_size;

    const float* x     = (const float*)d_in[0];   // [B,T,D]
    const float* Wqkv  = (const float*)d_in[1];   // [D, QKV_OUT]
    const float* bqkv  = (const float*)d_in[2];   // [QKV_OUT]
    const float* Wmlp  = (const float*)d_in[3];   // [HID, D]
    const float* bmlp  = (const float*)d_in[4];   // [D]
    float* out = (float*)d_out;

    char* ws = (char*)d_ws;
    size_t off = 0;
    auto alloc = [&](size_t bytes) {
        char* p = ws + off;
        off = (off + bytes + 255) & ~(size_t)255;
        return p;
    };
    unsigned short* xln    = (unsigned short*)alloc((size_t)ROWS * D_MODEL * 2);
    unsigned short* wqkv_t = (unsigned short*)alloc((size_t)D_MODEL * QKV_OUT * 2);
    unsigned short* wmlp_t = (unsigned short*)alloc((size_t)HID_DIM * D_MODEL * 2);
    float*          qkvh   = (float*)         alloc((size_t)ROWS * QKV_OUT * 4);
    unsigned short* qhb    = (unsigned short*)alloc((size_t)ROWS * D_MODEL * 2);
    unsigned short* khb    = (unsigned short*)alloc((size_t)ROWS * D_MODEL * 2);
    unsigned short* vhb    = (unsigned short*)alloc((size_t)ROWS * D_MODEL * 2);
    unsigned short* hid    = (unsigned short*)alloc((size_t)ROWS * HID_DIM * 2);

    // 1) LayerNorm(x) -> bf16
    ln_cast_x<<<ROWS, 256, 0, stream>>>(x, xln);

    // 2) weight transpose-casts: W[K][N] -> Wt[N][K] bf16
    {
        dim3 g1(QKV_OUT / 32, D_MODEL / 32);
        cast_transpose<<<g1, 256, 0, stream>>>(Wqkv, wqkv_t, D_MODEL, QKV_OUT);
        dim3 g2(D_MODEL / 32, HID_DIM / 32);
        cast_transpose<<<g2, 256, 0, stream>>>(Wmlp, wmlp_t, HID_DIM, D_MODEL);
    }

    // 3) QKV+hidden GEMM: [4096x2048] x [2048x14336] + b_qkv
    {
        dim3 grid(QKV_OUT / BN, ROWS / BM);
        gemm_bf16<false><<<grid, 256, 0, stream>>>(xln, wqkv_t, bqkv, nullptr,
                                                   qkvh, QKV_OUT, D_MODEL);
    }

    // 4) head LN (q,k) + pack (v) into [B,H,T,128] bf16
    {
        const int blocks = (BATCH * SEQ * NUM_HEADS) / 4;
        head_pack<<<blocks, 128, 0, stream>>>(qkvh, qhb, 0 * D_MODEL, 1);
        head_pack<<<blocks, 128, 0, stream>>>(qkvh, khb, 1 * D_MODEL, 1);
        head_pack<<<blocks, 128, 0, stream>>>(qkvh, vhb, 2 * D_MODEL, 0);
    }

    // 5) GELU(hidden) -> bf16
    gelu_cast<<<4096, 256, 0, stream>>>(qkvh, hid);

    // 6) MLP GEMM with bias + residual, writes full d_out
    {
        dim3 grid(D_MODEL / BN, ROWS / BM);
        gemm_bf16<true><<<grid, 256, 0, stream>>>(hid, wmlp_t, bmlp, x, out,
                                                  D_MODEL, HID_DIM);
    }

    // 7) flash attention accumulates into d_out
    attention<<<BATCH * NUM_HEADS * (SEQ / 128), 256, 0, stream>>>(qhb, khb, vhb, out);
}
